// TemporalGAT_71717363909218
// MI455X (gfx1250) — matrix operand order repaired and checked by reading, not compiled, not run on hardware
//
#include <hip/hip_runtime.h>
#include <hip/hip_bf16.h>
#include <math.h>

// ---------------------------------------------------------------------------
// TemporalGAT forward for MI455X (gfx1250, wave32).
// Dense GEMMs: v_wmma_f32_16x16x32_f16 (f16 in, f32 acc), all operands moved
// as b128 transactions (LDS-staged A tile, pre-transposed f16 weights).
// Edge softmax / aggregation: L2-resident scatter with float atomics.
// ---------------------------------------------------------------------------

typedef __attribute__((ext_vector_type(16))) _Float16 v16h;
typedef __attribute__((ext_vector_type(8)))  _Float16 v8h;
typedef __attribute__((ext_vector_type(4)))  _Float16 v4h;
typedef __attribute__((ext_vector_type(8)))  float    v8f;

#define FEATS 128

// ---------------- helpers ----------------

__device__ inline float warp_allsum(float v) {
  #pragma unroll
  for (int o = 16; o > 0; o >>= 1) v += __shfl_xor(v, o);
  return v;
}

// Correct float atomic-max via ordered integer punning.
__device__ inline void atomicMaxFloat(float* addr, float v) {
  if (v >= 0.0f) atomicMax((int*)addr, __float_as_int(v));
  else           atomicMin((unsigned int*)addr, __float_as_uint(v));
}

// ---------------- elementwise utility kernels ----------------

// 4 elements per thread: float4 load -> 4xf16 (b64) store.
__global__ void cvt_f32_to_f16(const float* __restrict__ in,
                               _Float16* __restrict__ out, int n) {
  const int i4 = (blockIdx.x * blockDim.x + threadIdx.x) * 4;
  if (i4 + 3 < n) {
    const float4 v = *(const float4*)(in + i4);
    v4h o; o[0] = (_Float16)v.x; o[1] = (_Float16)v.y;
           o[2] = (_Float16)v.z; o[3] = (_Float16)v.w;
    *(v4h*)(out + i4) = o;
  } else {
    for (int i = i4; i < n; ++i) out[i] = (_Float16)in[i];
  }
}

// Convert + transpose weights: in[k][n] (KxN row-major f32) -> out[n][k] f16.
__global__ void cvt_transpose_f16(const float* __restrict__ in,
                                  _Float16* __restrict__ out, int K, int Ncol) {
  const int i = blockIdx.x * blockDim.x + threadIdx.x;
  if (i >= K * Ncol) return;
  const int k = i / Ncol, n = i - k * Ncol;
  out[(size_t)n * K + k] = (_Float16)in[i];
}

// float4-vectorized fill.
__global__ void fill_f32(float* __restrict__ p, float v, int n) {
  const int i4 = (blockIdx.x * blockDim.x + threadIdx.x) * 4;
  if (i4 + 3 < n) {
    float4 o; o.x = v; o.y = v; o.z = v; o.w = v;
    *(float4*)(p + i4) = o;
  } else {
    for (int i = i4; i < n; ++i) p[i] = v;
  }
}

// ---------------- WMMA GEMM: C[M x 128] = A[M x 128] * B[128 x 128] --------
// A row-major f16; Bt is B transposed (Bt[n][k] = B[k][n]) f16; C row-major f32.
// Grid.x = M/16; block = 256 threads = 8 waves; wave w -> cols [16w, 16w+16).
// Register layouts per CDNA5 ISA 7.12.2:
//   A (16x32 f16): lane 0-15 row M=laneIdx; VGPR0-3 hold K = 8*laneHalf + 0..7,
//                  VGPR4-7 hold same K + 16.
//   B (32x16 f16): lane col N=laneIdx; VGPR0-3 hold K = 8*laneHalf + 0..7,
//                  VGPR4-7 hold same K + 16 (same order as A).
//   C/D (16x16 f32): VGPR r -> row (r + 8*laneHalf), col laneIdx.
__global__ void gemm_wmma_f16(const _Float16* __restrict__ A,
                              const _Float16* __restrict__ Bt,
                              float* __restrict__ C, int M) {
  __shared__ __align__(16) _Float16 As[16][136];  // +8 pad breaks bank pattern

  const int tileM = blockIdx.x;
  const int tid = threadIdx.x;

  // Cooperative A-tile stage: 16 rows x 128 f16 = 4KB; 256 threads x 16B (b128).
  {
    const int r = tid >> 4;
    const int c = (tid & 15) * 8;
    *(v8h*)&As[r][c] = *(const v8h*)(A + (size_t)(tileM * 16 + r) * FEATS + c);
  }
  __syncthreads();

  const int wave = tid >> 5;
  const int lane = tid & 31;
  const int laneHalf = lane >> 4;
  const int laneIdx  = lane & 15;
  const int col = wave * 16 + laneIdx;

  v8f acc = {};
  #pragma unroll
  for (int kc = 0; kc < 4; ++kc) {            // K = 128 in chunks of 32
    // A fragment from LDS: two ds_load_b128
    const int ka = kc * 32 + laneHalf * 8;
    const v8h alo = *(const v8h*)&As[laneIdx][ka];
    const v8h ahi = *(const v8h*)&As[laneIdx][ka + 16];
    const v16h a = __builtin_shufflevector(alo, ahi,
        0,1,2,3,4,5,6,7,8,9,10,11,12,13,14,15);
    // B fragment from transposed weights: two global_load_b128 (K and K+16)
    const int kb = kc * 32 + laneHalf * 8;
    const _Float16* bp = Bt + (size_t)col * FEATS + kb;
    const v8h blo = *(const v8h*)bp;
    const v8h bhi = *(const v8h*)(bp + 16);
    const v16h b = __builtin_shufflevector(blo, bhi,
        0,1,2,3,4,5,6,7,8,9,10,11,12,13,14,15);

    acc = __builtin_amdgcn_wmma_f32_16x16x32_f16(
        /*neg_a=*/false, a, /*neg_b=*/false, b,
        /*c_mod=*/(short)0, acc, /*reuse_a=*/false, /*reuse_b=*/false);
  }

  float* cp = C + (size_t)(tileM * 16 + laneHalf * 8) * FEATS + col;
  #pragma unroll
  for (int r = 0; r < 8; ++r) cp[(size_t)r * FEATS] = acc[r];
  (void)M;
}

// ---------------- attention score dots: el/er [N x H] ----------------------
__global__ void scores_kernel(const float* __restrict__ f,
                              const float* __restrict__ al,
                              const float* __restrict__ ar,
                              float* __restrict__ el, float* __restrict__ er,
                              int N, int H) {
  const int n = blockIdx.x * (blockDim.x >> 5) + (threadIdx.x >> 5);
  const int lane = threadIdx.x & 31;
  if (n >= N) return;
  const int D = FEATS / H;
  float a0 = 0.f, a1 = 0.f, b0 = 0.f, b1 = 0.f;
  #pragma unroll
  for (int t = 0; t < 4; ++t) {
    const int j = lane + 32 * t;
    const float v = f[(size_t)n * FEATS + j];
    const float pa = v * al[j], pb = v * ar[j];
    if (j < D) { a0 += pa; b0 += pb; } else { a1 += pa; b1 += pb; }
  }
  a0 = warp_allsum(a0); b0 = warp_allsum(b0);
  a1 = warp_allsum(a1); b1 = warp_allsum(b1);
  if (lane == 0) {
    el[(size_t)n * H] = a0; er[(size_t)n * H] = b0;
    if (H == 2) { el[(size_t)n * H + 1] = a1; er[(size_t)n * H + 1] = b1; }
  }
}

// ---------------- edge pass 1: e = leaky_relu(el[src]+er[dst]); segment max -
__global__ void edge_max_kernel(const int* __restrict__ src, const int* __restrict__ dst,
                                const float* __restrict__ el, const float* __restrict__ er,
                                float* __restrict__ e_edge, float* __restrict__ mmax,
                                int E, int H) {
  const int i = blockIdx.x * blockDim.x + threadIdx.x;
  if (i >= E * H) return;
  const int eid = i / H, h = i - eid * H;
  const int s = src[eid], d = dst[eid];
  float e = el[(size_t)s * H + h] + er[(size_t)d * H + h];
  e = (e > 0.f) ? e : 0.2f * e;              // leaky relu, slope 0.2
  e_edge[i] = e;
  atomicMaxFloat(&mmax[(size_t)d * H + h], e);
}

// ---------------- edge pass 2: ex = exp(e - m[dst]); segment sum -----------
__global__ void edge_exp_kernel(const int* __restrict__ dst,
                                float* __restrict__ e_edge,
                                const float* __restrict__ mmax,
                                float* __restrict__ ssum, int E, int H) {
  const int i = blockIdx.x * blockDim.x + threadIdx.x;
  if (i >= E * H) return;
  const int eid = i / H, h = i - eid * H;
  const int d = dst[eid];
  const float ex = __expf(e_edge[i] - mmax[(size_t)d * H + h]);
  e_edge[i] = ex;
  atomicAdd(&ssum[(size_t)d * H + h], ex);
}

// ---------------- edge pass 3: rst[dst] += f[src] * alpha ------------------
// One wave32 per edge; lane gathers float4 of f[src] (b128) + 4 atomic adds.
__global__ void edge_aggr_kernel(const int* __restrict__ src, const int* __restrict__ dst,
                                 const float* __restrict__ f,
                                 const float* __restrict__ e_edge,
                                 const float* __restrict__ ssum,
                                 float* __restrict__ rst, int E, int H) {
  const int eid = blockIdx.x * (blockDim.x >> 5) + (threadIdx.x >> 5);
  const int lane = threadIdx.x & 31;
  if (eid >= E) return;
  const int s = src[eid], d = dst[eid];
  const int j0 = lane * 4;                    // features [j0, j0+3], same head
  const int h = (H == 2) ? (j0 >> 6) : 0;     // D=64 -> head = j0/64
  const float alpha = e_edge[(size_t)eid * H + h] / ssum[(size_t)d * H + h];
  const float4 v = *(const float4*)(f + (size_t)s * FEATS + j0);
  float* rp = rst + (size_t)d * FEATS + j0;
  atomicAdd(rp + 0, v.x * alpha);
  atomicAdd(rp + 1, v.y * alpha);
  atomicAdd(rp + 2, v.z * alpha);
  atomicAdd(rp + 3, v.w * alpha);
}

// ---------------- fused bias + layernorm + elu (layer 1 epilogue) ----------
__global__ void ln_elu_kernel(const float* __restrict__ rst,
                              const float* __restrict__ bias,
                              const float* __restrict__ g,
                              const float* __restrict__ be,
                              float* __restrict__ hout,
                              _Float16* __restrict__ hh, int N) {
  const int n = blockIdx.x * (blockDim.x >> 5) + (threadIdx.x >> 5);
  const int lane = threadIdx.x & 31;
  if (n >= N) return;
  float v[4]; float s = 0.f;
  #pragma unroll
  for (int t = 0; t < 4; ++t) {
    const int j = lane + 32 * t;
    v[t] = rst[(size_t)n * FEATS + j] + bias[j];
    s += v[t];
  }
  const float mu = warp_allsum(s) * (1.f / FEATS);
  float q = 0.f;
  #pragma unroll
  for (int t = 0; t < 4; ++t) { const float d = v[t] - mu; q += d * d; }
  const float rs = rsqrtf(warp_allsum(q) * (1.f / FEATS) + 1e-5f);
  #pragma unroll
  for (int t = 0; t < 4; ++t) {
    const int j = lane + 32 * t;
    float y = (v[t] - mu) * rs * g[j] + be[j];
    y = (y > 0.f) ? y : (__expf(y) - 1.f);   // ELU
    hout[(size_t)n * FEATS + j] = y;
    hh[(size_t)n * FEATS + j] = (_Float16)y;
  }
}

// ---------------- final: LN(rst2 + b2 + h + (x@Wp) + bp) -------------------
__global__ void final_ln_kernel(const float* __restrict__ rst,
                                const float* __restrict__ b2,
                                const float* __restrict__ h,
                                const float* __restrict__ hres,
                                const float* __restrict__ bp,
                                const float* __restrict__ g,
                                const float* __restrict__ be,
                                float* __restrict__ out, int N) {
  const int n = blockIdx.x * (blockDim.x >> 5) + (threadIdx.x >> 5);
  const int lane = threadIdx.x & 31;
  if (n >= N) return;
  float v[4]; float s = 0.f;
  #pragma unroll
  for (int t = 0; t < 4; ++t) {
    const int j = lane + 32 * t;
    v[t] = rst[(size_t)n * FEATS + j] + b2[j] + h[(size_t)n * FEATS + j]
         + hres[(size_t)n * FEATS + j] + bp[j];
    s += v[t];
  }
  const float mu = warp_allsum(s) * (1.f / FEATS);
  float q = 0.f;
  #pragma unroll
  for (int t = 0; t < 4; ++t) { const float d = v[t] - mu; q += d * d; }
  const float rs = rsqrtf(warp_allsum(q) * (1.f / FEATS) + 1e-5f);
  #pragma unroll
  for (int t = 0; t < 4; ++t) {
    const int j = lane + 32 * t;
    out[(size_t)n * FEATS + j] = (v[t] - mu) * rs * g[j] + be[j];
  }
}

// ---------------------------------------------------------------------------

static inline size_t alignUp(size_t x) { return (x + 255) & ~(size_t)255; }

extern "C" void kernel_launch(void* const* d_in, const int* in_sizes, int n_in,
                              void* d_out, int out_size, void* d_ws, size_t ws_size,
                              hipStream_t stream) {
  const float* x   = (const float*)d_in[0];
  const int*   src = (const int*)  d_in[1];
  const int*   dst = (const int*)  d_in[2];
  const float* W1  = (const float*)d_in[3];
  const float* al1 = (const float*)d_in[4];
  const float* ar1 = (const float*)d_in[5];
  const float* b1  = (const float*)d_in[6];
  const float* W2  = (const float*)d_in[7];
  const float* al2 = (const float*)d_in[8];
  const float* ar2 = (const float*)d_in[9];
  const float* b2  = (const float*)d_in[10];
  const float* Wp  = (const float*)d_in[11];
  const float* bp  = (const float*)d_in[12];
  const float* g1  = (const float*)d_in[13];
  const float* be1 = (const float*)d_in[14];
  const float* g2  = (const float*)d_in[15];
  const float* be2 = (const float*)d_in[16];
  float* out = (float*)d_out;

  const int N = in_sizes[0] / FEATS;   // 50000
  const int E = in_sizes[1];           // 800000
  const size_t NB = (size_t)N * FEATS;

  // ---- workspace layout ----
  char* ws = (char*)d_ws;
  size_t off = 0;
  _Float16* xh  = (_Float16*)(ws + off); off = alignUp(off + NB * 2);      // f16 acts
  _Float16* W1t = (_Float16*)(ws + off); off = alignUp(off + 16384 * 2);   // transposed f16
  _Float16* W2t = (_Float16*)(ws + off); off = alignUp(off + 16384 * 2);
  _Float16* Wpt = (_Float16*)(ws + off); off = alignUp(off + 16384 * 2);
  float* f    = (float*)(ws + off); off = alignUp(off + NB * 4);           // f1 then f2
  float* rst  = (float*)(ws + off); off = alignUp(off + NB * 4);           // aggregation
  float* hres = (float*)(ws + off); off = alignUp(off + NB * 4);           // x@Wp
  float* hbuf = (float*)(ws + off); off = alignUp(off + NB * 4);           // h after LN1
  float* el   = (float*)(ws + off); off = alignUp(off + (size_t)N * 2 * 4);
  float* er   = (float*)(ws + off); off = alignUp(off + (size_t)N * 2 * 4);
  float* mm   = (float*)(ws + off); off = alignUp(off + (size_t)N * 2 * 4);
  float* ss   = (float*)(ws + off); off = alignUp(off + (size_t)N * 2 * 4);
  float* e_edge = (float*)(ws + off); off = alignUp(off + (size_t)E * 2 * 4);
  (void)ws_size; (void)n_in; (void)out_size;

  const int TPB = 256;
  const float NEG_HUGE = -3.402823466e+38f;

  // ---- conversions (x -> f16; weights -> transposed f16) ----
  cvt_f32_to_f16<<<(int)((NB / 4 + TPB - 1) / TPB), TPB, 0, stream>>>(x, xh, (int)NB);
  cvt_transpose_f16<<<(16384 + TPB - 1) / TPB, TPB, 0, stream>>>(W1, W1t, FEATS, FEATS);
  cvt_transpose_f16<<<(16384 + TPB - 1) / TPB, TPB, 0, stream>>>(W2, W2t, FEATS, FEATS);
  cvt_transpose_f16<<<(16384 + TPB - 1) / TPB, TPB, 0, stream>>>(Wp, Wpt, FEATS, FEATS);

  const int mTiles = N / 16;  // 3125

  // ---- residual projection: hres = x @ Wp (bp folded into final LN) ----
  gemm_wmma_f16<<<mTiles, 256, 0, stream>>>(xh, Wpt, hres, N);

  // ================= GAT layer 1 (H=2, D=64) =================
  gemm_wmma_f16<<<mTiles, 256, 0, stream>>>(xh, W1t, f, N);            // f1 = x @ W1
  scores_kernel<<<(N + 7) / 8, 256, 0, stream>>>(f, al1, ar1, el, er, N, 2);

  fill_f32<<<(N * 2 / 4 + TPB - 1) / TPB, TPB, 0, stream>>>(mm, NEG_HUGE, N * 2);
  fill_f32<<<(N * 2 / 4 + TPB - 1) / TPB, TPB, 0, stream>>>(ss, 0.f, N * 2);
  fill_f32<<<(int)((NB / 4 + TPB - 1) / TPB), TPB, 0, stream>>>(rst, 0.f, (int)NB);

  edge_max_kernel<<<(E * 2 + TPB - 1) / TPB, TPB, 0, stream>>>(src, dst, el, er, e_edge, mm, E, 2);
  edge_exp_kernel<<<(E * 2 + TPB - 1) / TPB, TPB, 0, stream>>>(dst, e_edge, mm, ss, E, 2);
  edge_aggr_kernel<<<(E + 7) / 8, 256, 0, stream>>>(src, dst, f, e_edge, ss, rst, E, 2);

  // h = elu(LN(rst + b1)); emits f16 copy into xh for GEMM2
  ln_elu_kernel<<<(N + 7) / 8, 256, 0, stream>>>(rst, b1, g1, be1, hbuf, xh, N);

  // ================= GAT layer 2 (H=1, D=128) =================
  gemm_wmma_f16<<<mTiles, 256, 0, stream>>>(xh, W2t, f, N);            // f2 = h @ W2
  scores_kernel<<<(N + 7) / 8, 256, 0, stream>>>(f, al2, ar2, el, er, N, 1);

  fill_f32<<<(N / 4 + TPB - 1) / TPB, TPB, 0, stream>>>(mm, NEG_HUGE, N);
  fill_f32<<<(N / 4 + TPB - 1) / TPB, TPB, 0, stream>>>(ss, 0.f, N);
  fill_f32<<<(int)((NB / 4 + TPB - 1) / TPB), TPB, 0, stream>>>(rst, 0.f, (int)NB);

  edge_max_kernel<<<(E + TPB - 1) / TPB, TPB, 0, stream>>>(src, dst, el, er, e_edge, mm, E, 1);
  edge_exp_kernel<<<(E + TPB - 1) / TPB, TPB, 0, stream>>>(dst, e_edge, mm, ss, E, 1);
  edge_aggr_kernel<<<(E + 7) / 8, 256, 0, stream>>>(src, dst, f, e_edge, ss, rst, E, 1);

  // out = LN(rst2 + b2 + h + hres + bp)
  final_ln_kernel<<<(N + 7) / 8, 256, 0, stream>>>(rst, b2, hbuf, hres, bp, g2, be2, out, N);
}
